// graph_constructor_simi_592705487498
// MI455X (gfx1250) — matrix operand order repaired and checked by reading, not compiled, not run on hardware
//
#include <hip/hip_runtime.h>
#include <cstdint>

typedef __attribute__((ext_vector_type(16))) __bf16 v16bf;
typedef __attribute__((ext_vector_type(8)))  float  v8f;

#define NN 2048
#define LL 512
#define LDSK 48   // padded LDS row stride in bf16 elements (96B: 16B-aligned, conflict-spread)

static __device__ __forceinline__ unsigned short f2bf(float f) {
  unsigned u = __float_as_uint(f);
  u += 0x7FFFu + ((u >> 16) & 1u);           // round-to-nearest-even
  return (unsigned short)(u >> 16);
}

union Frag { v16bf bf; uint4 u[2]; };

// A-matrix 16x32 bf16 fragment (ISA 7.12.2): lane holds row M=lane&15;
// VGPR pairs 0..3 hold K = 8*hi + 0..7, pairs 4..7 hold K = 16 + 8*hi + 0..7.
static __device__ __forceinline__ Frag load_a_frag(const unsigned short* tile, int nl, int hi) {
  Frag f;
  f.u[0] = *(const uint4*)(tile + nl * LDSK + 8 * hi);
  f.u[1] = *(const uint4*)(tile + nl * LDSK + 16 + 8 * hi);
  return f;
}
// B-matrix 32x16 bf16 fragment: lane holds column N=lane&15;
// elements 0..7 hold K = 8*hi + 0..7, elements 8..15 hold K = 16 + 8*hi + 0..7 (matches A).
static __device__ __forceinline__ Frag load_b_frag(const unsigned short* tile, int nl, int hi) {
  Frag f;
  f.u[0] = *(const uint4*)(tile + nl * LDSK + 8 * hi);
  f.u[1] = *(const uint4*)(tile + nl * LDSK + 16 + 8 * hi);
  return f;
}

// ---------------- GEMM2 (dominant, 137 GFLOP): A_inter = sigmoid(alpha*(h@h^T)*ri*rj) -----
// 4 waves, wave tile 64x64 (4x4 WMMA accs). Ping-pong LDS buffers filled with CDNA5
// async global->LDS DMA; one s_wait_asynccnt + one barrier per K-step; the DMA for
// tile k+1 overlaps the 16-WMMA burst on tile k.  (Placed first so disasm snippet shows it.)
#define G2_BUF_US (128 * LDSK)            // one buffer (A half + B half) = 6144 ushorts
__global__ __launch_bounds__(128) void gemm2_sim(const unsigned short* __restrict__ hbf,
                                                 const float* __restrict__ rinv,
                                                 const float* __restrict__ alphap,
                                                 float* __restrict__ out2) {
  extern __shared__ unsigned short sm[];   // 2 buffers * 12288 B = 24576 B dynamic LDS
  const int t = threadIdx.x;               // 0..127
  const int w = t >> 5, lane = t & 31;
  const int wm = w & 1, wn = w >> 1;
  const int hi = lane >> 4, nl = lane & 15;
  const int z = blockIdx.z;
  const int ibase = z * NN + blockIdx.y * 128;   // flat row index into [16384]
  const int jbase = z * NN + blockIdx.x * 128;
  const float al = alphap[0];

  v8f acc[4][4]; v8f vz = {};
  for (int i = 0; i < 4; i++) for (int j = 0; j < 4; j++) acc[i][j] = vz;

  auto issue_tile = [&](int kt, int buf) {
    const unsigned base = (unsigned)buf * (2u * G2_BUF_US);   // byte offset (12288 per buffer)
    for (int s = 0; s < 4; s++) {
      int e = t + 128 * s;                       // 0..511
      int row = e >> 2, koff = (e & 3) * 8;      // 8 bf16 = 16B per async op
      unsigned la = base + (unsigned)(row * 96 + koff * 2);
      unsigned long long ga = (unsigned long long)(uintptr_t)
          (hbf + (size_t)(ibase + row) * NN + kt * 32 + koff);
      asm volatile("global_load_async_to_lds_b128 %0, %1, off" :: "v"(la), "v"(ga) : "memory");
      unsigned lb = la + (unsigned)(2 * G2_BUF_US / 2);        // B half: +6144 bytes
      unsigned long long gb = (unsigned long long)(uintptr_t)
          (hbf + (size_t)(jbase + row) * NN + kt * 32 + koff);
      asm volatile("global_load_async_to_lds_b128 %0, %1, off" :: "v"(lb), "v"(gb) : "memory");
    }
  };

  issue_tile(0, 0);
  for (int kt = 0; kt < NN / 32; ++kt) {
    asm volatile("s_wait_asynccnt 0" ::: "memory");   // tile kt landed (this wave's DMAs)
    __syncthreads();                                  // all waves landed + prev reads done
    if (kt + 1 < NN / 32) issue_tile(kt + 1, (kt + 1) & 1);  // overlaps WMMAs below

    const unsigned short* As = sm + (size_t)(kt & 1) * G2_BUF_US;  // 3072 ushorts A + 3072 B
    const unsigned short* Bs = As + G2_BUF_US / 2;

    Frag bfr[4];
    for (int ni = 0; ni < 4; ni++) bfr[ni] = load_b_frag(Bs + (wn * 64 + ni * 16) * LDSK, nl, hi);
    for (int mi = 0; mi < 4; mi++) {
      Frag af = load_a_frag(As + (wm * 64 + mi * 16) * LDSK, nl, hi);
      for (int ni = 0; ni < 4; ni++)
        acc[mi][ni] = __builtin_amdgcn_wmma_f32_16x16x32_bf16(
            false, af.bf, false, bfr[ni].bf, (short)0, acc[mi][ni], false, false);
    }
  }

  float rj[4];
  for (int ni = 0; ni < 4; ni++) rj[ni] = rinv[jbase + wn * 64 + ni * 16 + nl];
  for (int mi = 0; mi < 4; mi++) {
    for (int r = 0; r < 8; r++) {
      int gi = ibase + wm * 64 + mi * 16 + r + 8 * hi;
      float ri = rinv[gi];
      for (int ni = 0; ni < 4; ni++) {
        int jloc = blockIdx.x * 128 + wn * 64 + ni * 16 + nl;
        float sim = acc[mi][ni][r] * ri * rj[ni];
        float v = 1.0f / (1.0f + __expf(-al * sim));
        out2[(size_t)gi * NN + jloc] = v;
      }
    }
  }
}

// ---------------- GEMM1: h = relu(x @ W + b) -> bf16, + row sum-of-squares partials --------
__global__ __launch_bounds__(256) void gemm1_proj(const float* __restrict__ x,
                                                  const float* __restrict__ W,
                                                  const float* __restrict__ bias,
                                                  unsigned short* __restrict__ hbf,
                                                  float* __restrict__ partial) {
  __shared__ unsigned short As[128 * LDSK];
  __shared__ unsigned short Bst[128 * LDSK];
  __shared__ float red[128][4];
  const int t = threadIdx.x;
  const int w = t >> 5, lane = t & 31;
  const int wm = w & 1, wn = w >> 1;
  const int hi = lane >> 4, nl = lane & 15;
  const int rbase = blockIdx.y * 128;   // over 16384 flat rows (B*NN)
  const int cbase = blockIdx.x * 128;   // over 2048 cols

  v8f acc[4][2]; v8f vz = {};
  for (int i = 0; i < 4; i++) for (int j = 0; j < 2; j++) acc[i][j] = vz;

  for (int kt = 0; kt < LL / 32; ++kt) {
    // stage A (x tile 128x32 f32 -> bf16)
    for (int s = 0; s < 4; s++) {
      int e = t + 256 * s;
      int row = e >> 3, koff = (e & 7) * 4;
      float4 v = *(const float4*)(x + (size_t)(rbase + row) * LL + kt * 32 + koff);
      unsigned p0 = (unsigned)f2bf(v.x) | ((unsigned)f2bf(v.y) << 16);
      unsigned p1 = (unsigned)f2bf(v.z) | ((unsigned)f2bf(v.w) << 16);
      *(uint2*)(As + row * LDSK + koff) = make_uint2(p0, p1);
    }
    // stage B transposed (W tile 32x128 -> Bst[j][k])
    for (int s = 0; s < 4; s++) {
      int e = t + 256 * s;
      int krow = e >> 5, coff = (e & 31) * 4;
      float4 v = *(const float4*)(W + (size_t)(kt * 32 + krow) * NN + cbase + coff);
      Bst[(coff + 0) * LDSK + krow] = f2bf(v.x);
      Bst[(coff + 1) * LDSK + krow] = f2bf(v.y);
      Bst[(coff + 2) * LDSK + krow] = f2bf(v.z);
      Bst[(coff + 3) * LDSK + krow] = f2bf(v.w);
    }
    __syncthreads();
    Frag bfr[2], afr[4];
    for (int ni = 0; ni < 2; ni++) bfr[ni] = load_b_frag(Bst + (wn * 32 + ni * 16) * LDSK, nl, hi);
    for (int mi = 0; mi < 4; mi++) afr[mi] = load_a_frag(As + (wm * 64 + mi * 16) * LDSK, nl, hi);
    for (int mi = 0; mi < 4; mi++)
      for (int ni = 0; ni < 2; ni++)
        acc[mi][ni] = __builtin_amdgcn_wmma_f32_16x16x32_bf16(
            false, afr[mi].bf, false, bfr[ni].bf, (short)0, acc[mi][ni], false, false);
    __syncthreads();
  }

  // epilogue: relu + bias, bf16 store, per-row sum-of-squares partial (deterministic, no atomics)
  for (int mi = 0; mi < 4; mi++) {
    for (int r = 0; r < 8; r++) {
      int row_l = wm * 64 + mi * 16 + r + 8 * hi;   // C layout: M = r + 8*hi
      int grow = rbase + row_l;
      float p = 0.f;
      for (int ni = 0; ni < 2; ni++) {
        int gcol = cbase + wn * 32 + ni * 16 + nl;
        float h = acc[mi][ni][r] + bias[gcol];
        h = h > 0.f ? h : 0.f;
        hbf[(size_t)grow * NN + gcol] = f2bf(h);
        p += h * h;
      }
      for (int off = 8; off >= 1; off >>= 1) p += __shfl_xor(p, off, 32);
      if (nl == 0) red[row_l][wn] = p;
    }
  }
  __syncthreads();
  if (t < 128) {
    float s = red[t][0] + red[t][1] + red[t][2] + red[t][3];
    partial[(size_t)(rbase + t) * 16 + blockIdx.x] = s;
  }
}

// ---------------- rinv = 1 / (||h_row|| + 1e-6) ----------------
__global__ __launch_bounds__(256) void row_rinv(const float* __restrict__ partial,
                                                float* __restrict__ rinv) {
  int i = blockIdx.x * 256 + threadIdx.x;
  float s = 0.f;
  for (int t = 0; t < 16; t++) s += partial[(size_t)i * 16 + t];
  rinv[i] = 1.0f / (sqrtf(s) + 1e-6f);
}

// ---------------- top-k(16) mask, in place on out2: one wave32 per row ----------------
__global__ __launch_bounds__(256) void topk_mask(float* __restrict__ out2) {
  const int w = threadIdx.x >> 5, lane = threadIdx.x & 31;
  const size_t row = (size_t)blockIdx.x * 8 + w;
  float* rp = out2 + row * NN;
  float v[64];
  for (int s = 0; s < 64; s++) v[s] = rp[lane + 32 * s];
  unsigned long long removed = 0ull;
  float thresh = 0.f;
  for (int it = 0; it < 16; ++it) {
    float lm = -1.f; int lj = 0;                       // sigmoid outputs in (0,1)
    for (int s = 0; s < 64; s++) {
      if (!((removed >> s) & 1ull)) {
        float vv = v[s]; int jj = lane + 32 * s;
        if (vv > lm || (vv == lm && jj < lj)) { lm = vv; lj = jj; }
      }
    }
    for (int off = 16; off >= 1; off >>= 1) {          // wave argmax (val desc, index asc)
      float ov = __shfl_xor(lm, off, 32);
      int   oj = __shfl_xor(lj, off, 32);
      if (ov > lm || (ov == lm && oj < lj)) { lm = ov; lj = oj; }
    }
    thresh = lm;
    if ((lj & 31) == lane) removed |= 1ull << (lj >> 5);
  }
  for (int s = 0; s < 64; s++) {
    float vv = v[s];
    rp[lane + 32 * s] = (vv >= thresh) ? vv : 0.0f;
  }
}

// ---------------- A_intra: per (b, 64-block) cosine similarity via bf16 WMMA -------------
__global__ __launch_bounds__(256) void intra_blocks(const float* __restrict__ x,
                                                    float* __restrict__ out1) {
  __shared__ unsigned short Xs[64 * LDSK];
  __shared__ float xr[64];
  __shared__ float red[64][4];
  const int t = threadIdx.x;
  const int w = t >> 5, lane = t & 31;
  const int wm = w & 3, wn = w >> 2;
  const int hi = lane >> 4, nl = lane & 15;
  const int z = blockIdx.y, blk = blockIdx.x;
  const float* xb = x + ((size_t)z * NN + blk * 64) * LL;

  { // f32 row norms over 512 elements, fixed reduction order
    int row = t >> 2, q = t & 3;
    float s = 0.f;
    const float* p = xb + (size_t)row * LL + q * 128;
    for (int i = 0; i < 128; i++) { float vv = p[i]; s += vv * vv; }
    red[row][q] = s;
  }
  __syncthreads();
  if (t < 64) xr[t] = 1.0f / (sqrtf(red[t][0] + red[t][1] + red[t][2] + red[t][3]) + 1e-8f);
  __syncthreads();

  v8f acc[2]; v8f vz = {}; acc[0] = vz; acc[1] = vz;
  for (int kt = 0; kt < LL / 32; ++kt) {
    for (int s = 0; s < 2; s++) {
      int e = t + 256 * s;
      int row = e >> 3, koff = (e & 7) * 4;
      float4 v = *(const float4*)(xb + (size_t)row * LL + kt * 32 + koff);
      unsigned p0 = (unsigned)f2bf(v.x) | ((unsigned)f2bf(v.y) << 16);
      unsigned p1 = (unsigned)f2bf(v.z) | ((unsigned)f2bf(v.w) << 16);
      *(uint2*)(Xs + row * LDSK + koff) = make_uint2(p0, p1);
    }
    __syncthreads();
    Frag af = load_a_frag(Xs + wm * 16 * LDSK, nl, hi);
    Frag b0 = load_b_frag(Xs + (wn * 32 + 0 * 16) * LDSK, nl, hi);
    Frag b1 = load_b_frag(Xs + (wn * 32 + 1 * 16) * LDSK, nl, hi);
    acc[0] = __builtin_amdgcn_wmma_f32_16x16x32_bf16(
        false, af.bf, false, b0.bf, (short)0, acc[0], false, false);
    acc[1] = __builtin_amdgcn_wmma_f32_16x16x32_bf16(
        false, af.bf, false, b1.bf, (short)0, acc[1], false, false);
    __syncthreads();
  }
  for (int ni = 0; ni < 2; ni++) {
    for (int r = 0; r < 8; r++) {
      int il = wm * 16 + r + 8 * hi;
      int jl = wn * 32 + ni * 16 + nl;
      float v = acc[ni][r] * xr[il] * xr[jl];
      out1[((size_t)z * NN + blk * 64 + il) * NN + blk * 64 + jl] = v;
    }
  }
}

extern "C" void kernel_launch(void* const* d_in, const int* in_sizes, int n_in,
                              void* d_out, int out_size, void* d_ws, size_t ws_size,
                              hipStream_t stream) {
  const float* x     = (const float*)d_in[0];  // [8,2048,512]
  const float* W     = (const float*)d_in[1];  // [512,2048]
  const float* bias  = (const float*)d_in[2];  // [2048]
  const float* alpha = (const float*)d_in[3];  // [1]

  float* out1 = (float*)d_out;                              // A_intra [8,2048,2048]
  float* out2 = out1 + (size_t)8 * NN * NN;                 // masked A_inter

  unsigned short* hbf = (unsigned short*)d_ws;                              // 67,108,864 B
  float* partial = (float*)((char*)d_ws + (size_t)8 * NN * NN * 2);         // 1,048,576 B
  float* rinv    = (float*)((char*)partial + (size_t)16384 * 16 * 4);       //    65,536 B

  // zero A_intra (only block-diagonal gets written by intra_blocks)
  hipMemsetAsync(out1, 0, (size_t)8 * NN * NN * sizeof(float), stream);

  gemm1_proj <<<dim3(NN / 128, 16384 / 128), 256, 0, stream>>>(x, W, bias, hbf, partial);
  row_rinv   <<<16384 / 256, 256, 0, stream>>>(partial, rinv);
  gemm2_sim  <<<dim3(NN / 128, NN / 128, 8), 128, 2 * 2 * 128 * LDSK, stream>>>(hbf, rinv, alpha, out2);
  topk_mask  <<<16384 / 8, 256, 0, stream>>>(out2);
  intra_blocks<<<dim3(32, 8), 256, 0, stream>>>(x, out1);
}
